// HybridQLSTM_65481071403225
// MI455X (gfx1250) — compile-verified, matrix operand order checked by reading, not run on hardware
//
#include <hip/hip_runtime.h>
#include <hip/hip_bf16.h>
#include <stdint.h>

typedef __bf16 bf16_t;
typedef __attribute__((ext_vector_type(16))) __bf16 v16bf;
typedef __attribute__((ext_vector_type(8)))  __bf16 v8bf;
typedef __attribute__((ext_vector_type(8)))  float  v8f;
typedef __attribute__((ext_vector_type(4)))  float  v4f;

// exact parameter types for the async global->LDS builtin (from hipcc diag):
// arg0: int __vector_size(16) __device__(AS1) *   arg1: LDS (AS3) equivalent
typedef int v4i_vs __attribute__((vector_size(16)));
typedef __attribute__((address_space(1))) v4i_vs* gv4i_p;
typedef __attribute__((address_space(3))) v4i_vs* lv4i_p;

#define T_STEPS 256
#define BATCH   64
#define DIM     1024
#define HID     1024
#define KTOT    2048   // D + H
#define KC      64     // staged K-chunk (2 WMMA k-steps)
#define A_ELEMS 4096   // 64 rows x 64 k
#define B_ELEMS 8192   // 128 rows x 64 k
#define BUF_ELEMS (A_ELEMS + B_ELEMS)   // 12288 bf16 = 24 KB
#define GSTRIDE (32 * KC)               // B gate stride in LDS elements

#ifndef __has_builtin
#define __has_builtin(x) 0
#endif

#if __has_builtin(__builtin_amdgcn_global_load_async_to_lds_b128)
#define USE_ASYNC 1
#else
#define USE_ASYNC 0
#endif

__device__ __forceinline__ bf16_t f2bf(float f) {
    unsigned u = __builtin_bit_cast(unsigned, f);
    unsigned short s = (unsigned short)((u + 0x7FFFu + ((u >> 16) & 1u)) >> 16);
    return __builtin_bit_cast(bf16_t, s);
}

__device__ __forceinline__ float sigm(float x) {
    return 1.0f / (1.0f + __expf(-x));
}
__device__ __forceinline__ float tanh_fast(float x) {
    return 1.0f - 2.0f / (__expf(2.0f * x) + 1.0f);   // saturates at +-1
}

// One 16-byte global->LDS transfer (async path on gfx1250).
__device__ __forceinline__ void copy16(const bf16_t* g, bf16_t* l) {
#if USE_ASYNC
    __builtin_amdgcn_global_load_async_to_lds_b128((gv4i_p)g, (lv4i_p)l, 0, 0);
#else
    *(v8bf*)l = *(const v8bf*)g;
#endif
}

template <int N>
__device__ __forceinline__ void wait_async() {
#if USE_ASYNC
#if __has_builtin(__builtin_amdgcn_s_wait_asynccnt)
    __builtin_amdgcn_s_wait_asynccnt(N);
#else
    asm volatile("s_wait_asynccnt %0" :: "n"(N) : "memory");
#endif
#endif
}

// WMMA 16-bit operand fragment from LDS (ISA 7.12.2 layout):
// lane half h: elems 0..7 = k ofs 8h..8h+7 ; elems 8..15 = k ofs 16+8h..+7
__device__ __forceinline__ v16bf load_frag(const bf16_t* base, int half) {
    v8bf r0 = *(const v8bf*)(base + 8 * half);
    v8bf r1 = *(const v8bf*)(base + 16 + 8 * half);
    v16bf a;
#pragma unroll
    for (int e = 0; e < 8; ++e) { a[e] = r0[e]; a[8 + e] = r1[e]; }
    return a;
}

#define WMMA_BF16(A, B, C) \
    __builtin_amdgcn_wmma_f32_16x16x32_bf16(false, (A), false, (B), (short)0, (C), false, false)

// ---------------- one-time conversion kernels ----------------

// Wc[j][k]: gate-fused weights, bf16, row-major [4096][2048]; j = g*1024+n.
__global__ __launch_bounds__(256) void conv_w(
    const float* __restrict__ Wf, const float* __restrict__ Wi,
    const float* __restrict__ Wg, const float* __restrict__ Wo,
    bf16_t* __restrict__ Wc)
{
    size_t idx = ((size_t)blockIdx.x * 256 + threadIdx.x) * 8;
    size_t j = idx >> 11;
    size_t k = idx & 2047;
    int g = (int)(j >> 10);
    const float* s = (g == 0 ? Wf : g == 1 ? Wi : g == 2 ? Wg : Wo)
                     + ((j & 1023) * (size_t)KTOT + k);
    v4f a = *(const v4f*)s;
    v4f b = *(const v4f*)(s + 4);
    v8bf o;
    o[0] = f2bf(a.x); o[1] = f2bf(a.y); o[2] = f2bf(a.z); o[3] = f2bf(a.w);
    o[4] = f2bf(b.x); o[5] = f2bf(b.y); o[6] = f2bf(b.z); o[7] = f2bf(b.w);
    *(v8bf*)(Wc + idx) = o;
}

__global__ __launch_bounds__(256) void conv_x(
    const float* __restrict__ X, bf16_t* __restrict__ Xbf)
{
    size_t idx = ((size_t)blockIdx.x * 256 + threadIdx.x) * 8;
    v4f a = *(const v4f*)(X + idx);
    v4f b = *(const v4f*)(X + idx + 4);
    v8bf o;
    o[0] = f2bf(a.x); o[1] = f2bf(a.y); o[2] = f2bf(a.z); o[3] = f2bf(a.w);
    o[4] = f2bf(b.x); o[5] = f2bf(b.y); o[6] = f2bf(b.z); o[7] = f2bf(b.w);
    *(v8bf*)(Xbf + idx) = o;
}

__global__ __launch_bounds__(256) void init_state(
    bf16_t* __restrict__ hA, bf16_t* __restrict__ hB, float* __restrict__ c)
{
    int i = blockIdx.x * 256 + threadIdx.x;
    hA[i] = f2bf(0.0f);
    hB[i] = f2bf(0.0f);
    c[i]  = 0.0f;
}

// ---------------- recurrent step ----------------
// Block = 8 waves = 4 M-tiles x 2 N-tiles; each K-chunk (64) is staged to LDS
// exactly once per block (async, double buffered), then all waves feed WMMA
// from ds_load_b128. Grid: 32 blocks x 256 threads.
__global__ __launch_bounds__(256) void lstm_step(
    const bf16_t* __restrict__ Xbf, const bf16_t* __restrict__ Wc,
    const float* __restrict__ bfp, const float* __restrict__ bip,
    const float* __restrict__ bgp, const float* __restrict__ bop,
    const bf16_t* __restrict__ hprev, bf16_t* __restrict__ hnext,
    float* __restrict__ cbuf, float* __restrict__ out,
    float* __restrict__ hx, float* __restrict__ cx, int t)
{
    __shared__ bf16_t smem[2][BUF_ELEMS];   // 48 KB

    const int tid  = threadIdx.x;
    const int lane = tid & 31;
    const int wid  = tid >> 5;
    const int mt   = wid & 3;          // batch tile: rows mt*16..+15
    const int ntl  = wid >> 2;         // 0/1: which 16-col slab in this block
    const int half = lane >> 4;
    const int ln   = lane & 15;
    const int nbase = blockIdx.x * 32;            // column base within a gate
    const int ncol  = nbase + ntl * 16 + ln;      // hidden unit for B/epilogue

    const bf16_t* xrow0 = Xbf + (size_t)t * BATCH * DIM;   // batch row 0 of x_t

    // ---- cooperative async fill of one K-chunk into an LDS buffer ----
    // A: 64 rows x 64 k  (src rows stride 1024 for both x and h parts)
    // B: 128 rows (= 4 gates x 2 slabs x 16 cols) x 64 k
    auto fill = [&](bf16_t* sb, int kc) {
        const bf16_t* asrc = (kc < DIM) ? (xrow0 + kc) : (hprev + (kc - DIM));
        const bf16_t* bsrc = Wc + kc;
#pragma unroll
        for (int s = 0; s < 6; ++s) {
            int i = tid + (s << 8);
            if (i < 512) {                         // A transfers: 64 rows x 8 segs
                int r = i >> 3, seg = i & 7;
                copy16(asrc + (size_t)r * 1024 + seg * 8,
                       sb + r * KC + seg * 8);
            } else {                               // B transfers: 128 rows x 8 segs
                int i2 = i - 512;
                int rb = i2 >> 3, seg = i2 & 7;
                int g = rb >> 5;                   // gate
                int rem = rb & 31;                 // slab*16 + col
                size_t j = (size_t)g * HID + nbase + rem;
                copy16(bsrc + j * KTOT + seg * 8,
                       sb + A_ELEMS + rb * KC + seg * 8);
            }
        }
    };

    v8f acc0 = {}, acc1 = {}, acc2 = {}, acc3 = {};

    fill(smem[0], 0);                              // prologue: chunk 0

#pragma unroll 1
    for (int c = 0; c < KTOT / KC; ++c) {
        int kcn = (c + 1) * KC;
        if (kcn < KTOT) {
            fill(smem[(c + 1) & 1], kcn);          // prefetch next chunk
            wait_async<6>();                       // oldest 6 (chunk c) done
        } else {
            wait_async<0>();
        }
        __syncthreads();                           // whole block's chunk c in LDS

        const bf16_t* sb    = smem[c & 1];
        const bf16_t* arow  = sb + (mt * 16 + ln) * KC;
        const bf16_t* bbase = sb + A_ELEMS + (ntl * 16 + ln) * KC;

        // Load ALL fragments of the chunk into distinct registers first so the
        // scheduler can clause the ds_load_b128s and use staged s_wait_dscnt
        // instead of a full drain before every WMMA.
        v16bf A0  = load_frag(arow, half);
        v16bf A1  = load_frag(arow + 32, half);
        v16bf B00 = load_frag(bbase + 0 * GSTRIDE, half);
        v16bf B10 = load_frag(bbase + 1 * GSTRIDE, half);
        v16bf B20 = load_frag(bbase + 2 * GSTRIDE, half);
        v16bf B30 = load_frag(bbase + 3 * GSTRIDE, half);
        v16bf B01 = load_frag(bbase + 0 * GSTRIDE + 32, half);
        v16bf B11 = load_frag(bbase + 1 * GSTRIDE + 32, half);
        v16bf B21 = load_frag(bbase + 2 * GSTRIDE + 32, half);
        v16bf B31 = load_frag(bbase + 3 * GSTRIDE + 32, half);

        acc0 = WMMA_BF16(A0, B00, acc0);
        acc1 = WMMA_BF16(A0, B10, acc1);
        acc2 = WMMA_BF16(A0, B20, acc2);
        acc3 = WMMA_BF16(A0, B30, acc3);
        acc0 = WMMA_BF16(A1, B01, acc0);
        acc1 = WMMA_BF16(A1, B11, acc1);
        acc2 = WMMA_BF16(A1, B21, acc2);
        acc3 = WMMA_BF16(A1, B31, acc3);

        __syncthreads();                           // buffer may be refilled next iter
    }

    // Epilogue: C/D layout — VGPR v: lanes 0-15 hold M=v, lanes 16-31 hold M=v+8.
    float bf_v = bfp[ncol], bi_v = bip[ncol], bg_v = bgp[ncol], bo_v = bop[ncol];
#pragma unroll
    for (int v = 0; v < 8; ++v) {
        int brow = mt * 16 + (half << 3) + v;
        size_t sidx = (size_t)brow * HID + ncol;
        float f = sigm(acc0[v] + bf_v);
        float i = sigm(acc1[v] + bi_v);
        float g = tanh_fast(acc2[v] + bg_v);
        float o = sigm(acc3[v] + bo_v);
        float cc = f * cbuf[sidx] + i * g;
        float h  = o * tanh_fast(cc);
        cbuf[sidx]  = cc;
        hnext[sidx] = f2bf(h);
        out[((size_t)t * BATCH + brow) * HID + ncol] = h;
        hx[sidx] = h;   // final step's write survives => hx
        cx[sidx] = cc;  // final cx
    }
}

extern "C" void kernel_launch(void* const* d_in, const int* in_sizes, int n_in,
                              void* d_out, int out_size, void* d_ws, size_t ws_size,
                              hipStream_t stream) {
    const float* X   = (const float*)d_in[0];
    const float* Wf  = (const float*)d_in[1];
    const float* bfb = (const float*)d_in[2];
    const float* Wi  = (const float*)d_in[3];
    const float* bib = (const float*)d_in[4];
    const float* Wg  = (const float*)d_in[5];
    const float* bgb = (const float*)d_in[6];
    const float* Wo  = (const float*)d_in[7];
    const float* bob = (const float*)d_in[8];

    char* ws = (char*)d_ws;
    bf16_t* Wc  = (bf16_t*)ws;                                     // 16 MB
    bf16_t* Xbf = (bf16_t*)(ws + (size_t)16 * 1024 * 1024);        // 32 MB
    bf16_t* hA  = (bf16_t*)(ws + (size_t)48 * 1024 * 1024);        // 128 KB
    bf16_t* hB  = hA + (size_t)BATCH * HID;                        // 128 KB
    float*  cb  = (float*)(hB + (size_t)BATCH * HID);              // 256 KB

    float* out = (float*)d_out;
    float* hx  = out + (size_t)T_STEPS * BATCH * HID;
    float* cx  = hx + (size_t)BATCH * HID;

    conv_w<<<4096, 256, 0, stream>>>(Wf, Wi, Wg, Wo, Wc);
    conv_x<<<8192, 256, 0, stream>>>(X, Xbf);
    init_state<<<256, 256, 0, stream>>>(hA, hB, cb);

    for (int t = 0; t < T_STEPS; ++t) {
        const bf16_t* hp = (t & 1) ? hB : hA;
        bf16_t*       hn = (t & 1) ? hA : hB;
        lstm_step<<<32, 256, 0, stream>>>(Xbf, Wc, bfb, bib, bgb, bob,
                                          hp, hn, cb, out, hx, cx, t);
    }
}
